// SycophancyDetector_31250182045926
// MI455X (gfx1250) — matrix operand order repaired and checked by reading, not compiled, not run on hardware
//
#include <hip/hip_runtime.h>
#include <hip/hip_bf16.h>
#include <math.h>

// ---------------------------------------------------------------------------
// SycophancyDetector forward for MI455X (gfx1250, wave32, WMMA bf16 path).
// All GEMMs run through V_WMMA_F32_16X16X32_BF16 (bf16 in, f32 accum).
// LDS staging uses 128-bit global loads + 128-bit ds ops; B tiles are stored
// in WMMA fragment order so each lane's fragment is one contiguous 32B read.
// ---------------------------------------------------------------------------

static constexpr int DM  = 384;   // model dim
static constexpr int NH  = 4;     // heads
static constexpr int HDM = 96;    // head dim
static constexpr int D3  = 1152;  // 3*DM
static constexpr int DHG = 1536;  // DIM*HEADS (global block)

typedef __attribute__((ext_vector_type(16))) __bf16 v16bf;
typedef __attribute__((ext_vector_type(8)))  float  v8f;

union FragU { v16bf v; uint4 q[2]; };
union V8U  { uint4 q; unsigned short s[8]; };

__device__ __forceinline__ unsigned short f2bf(float f) {
  unsigned u = __float_as_uint(f);
  return (unsigned short)((u + 0x7FFFu + ((u >> 16) & 1u)) >> 16);
}
__device__ __forceinline__ float bf2f(unsigned short h) {
  return __uint_as_float(((unsigned)h) << 16);
}
__device__ __forceinline__ float sigf(float x)  { return 1.f / (1.f + __expf(-x)); }
__device__ __forceinline__ float geluf(float x) { return 0.5f * x * (1.f + erff(x * 0.7071067811865475f)); }

__device__ __forceinline__ void atomicMaxF(float* addr, float v) {
  unsigned int* u = (unsigned int*)addr;
  unsigned int old = *u;
  while (__uint_as_float(old) < v) {
    unsigned int assumed = old;
    old = atomicCAS(u, assumed, __float_as_uint(v));
    if (old == assumed) break;
  }
}

__device__ __forceinline__ float blockReduceSum(float v, float* sbuf) {
  int tid = threadIdx.x;
  sbuf[tid] = v; __syncthreads();
  for (int st = blockDim.x >> 1; st > 0; st >>= 1) {
    if (tid < st) sbuf[tid] += sbuf[tid + st];
    __syncthreads();
  }
  float r = sbuf[0]; __syncthreads();
  return r;
}

// ------------------------------- elementwise -------------------------------
__global__ void sk_copyf(const float* s, float* d, long long n) {
  long long i = (long long)blockIdx.x * 256 + threadIdx.x; if (i < n) d[i] = s[i];
}
__global__ void sk_fill(float* p, float v, long long n) {
  long long i = (long long)blockIdx.x * 256 + threadIdx.x; if (i < n) p[i] = v;
}
__global__ void sk_f2bf(const float* s, unsigned short* d, long long n) {
  long long i = (long long)blockIdx.x * 256 + threadIdx.x; if (i < n) d[i] = f2bf(s[i]);
}
__global__ void sk_gelu(const float* s, float* d, long long n) {
  long long i = (long long)blockIdx.x * 256 + threadIdx.x; if (i < n) d[i] = geluf(s[i]);
}
__global__ void sk_gelu2bf(const float* s, unsigned short* d, long long n) {
  long long i = (long long)blockIdx.x * 256 + threadIdx.x; if (i < n) d[i] = f2bf(geluf(s[i]));
}
__global__ void sk_relu(const float* s, float* d, long long n) {
  long long i = (long long)blockIdx.x * 256 + threadIdx.x; if (i < n) d[i] = fmaxf(s[i], 0.f);
}
__global__ void sk_skip(float* x, const float* o, const float* skp, long long n) {
  long long i = (long long)blockIdx.x * 256 + threadIdx.x;
  if (i < n) { float sk = sigf(*skp); x[i] = sk * o[i] + (1.f - sk) * x[i]; }
}
__global__ void sk_scalebf(const float* s, const float* sumsq, unsigned short* d, long long n) {
  long long i = (long long)blockIdx.x * 256 + threadIdx.x;
  if (i < n) d[i] = f2bf(s[i] * rsqrtf(*sumsq));
}
__global__ void sk_mix(const float* ai, const float* x1, const float* alphap, float* out, long long n) {
  long long i = (long long)blockIdx.x * 256 + threadIdx.x;
  if (i < n) { float a = sigf(*alphap); out[i] = (1.f - a) * ai[i] + a * x1[i]; }
}

// --------------------------- reductions / LN -------------------------------
__global__ __launch_bounds__(256) void sk_layernorm(const float* in, const float* g,
                                                    const float* b, float* out) {
  __shared__ float s1[256], s2[256];
  long long row = blockIdx.x;
  const float* x = in + row * DM;
  int tid = threadIdx.x;
  float s = 0.f, ss = 0.f;
  for (int c = tid; c < DM; c += 256) { float v = x[c]; s += v; ss += v * v; }
  s1[tid] = s; s2[tid] = ss; __syncthreads();
  for (int st = 128; st > 0; st >>= 1) {
    if (tid < st) { s1[tid] += s1[tid + st]; s2[tid] += s2[tid + st]; }
    __syncthreads();
  }
  float mean = s1[0] / DM;
  float var  = s2[0] / DM - mean * mean;
  float rstd = rsqrtf(var + 1e-5f);
  for (int c = tid; c < DM; c += 256) out[row * DM + c] = (x[c] - mean) * rstd * g[c] + b[c];
}

__global__ __launch_bounds__(256) void sk_sumsq(const float* x, float* out, long long n) {
  __shared__ float red[256];
  float s = 0.f;
  for (long long i = (long long)blockIdx.x * 256 + threadIdx.x; i < n; i += (long long)gridDim.x * 256) {
    float v = x[i]; s += v * v;
  }
  float r = blockReduceSum(s, red);
  if (!threadIdx.x) atomicAdd(out, r);
}
__global__ __launch_bounds__(256) void sk_colsum_bf(const unsigned short* x, float* out, int rows, int cols) {
  __shared__ float red[256];
  int c = blockIdx.x; float s = 0.f;
  for (int r = threadIdx.x; r < rows; r += 256) s += bf2f(x[(long long)r * cols + c]);
  float t = blockReduceSum(s, red);
  if (!threadIdx.x) out[c] = t;
}
__global__ __launch_bounds__(256) void sk_colmean(const float* x, float* out, int rows, int cols) {
  __shared__ float red[256];
  int c = blockIdx.x; float s = 0.f;
  for (int r = threadIdx.x; r < rows; r += 256) s += x[(long long)r * cols + c];
  float t = blockReduceSum(s, red);
  if (!threadIdx.x) out[c] = t / rows;
}
__global__ __launch_bounds__(256) void sk_colwsum(const float* x, const float* w, float* out, int rows, int cols) {
  __shared__ float red[256];
  int c = blockIdx.x; float s = 0.f;
  for (int r = threadIdx.x; r < rows; r += 256) s += x[(long long)r * cols + c] * w[r];
  float t = blockReduceSum(s, red);
  if (!threadIdx.x) out[c] = t;
}
__global__ __launch_bounds__(256) void sk_softmax_vec(const float* a, float* w, int n) {
  __shared__ float red[256];
  int tid = threadIdx.x;
  float mx = -1e30f;
  for (int i = tid; i < n; i += 256) mx = fmaxf(mx, a[i]);
  red[tid] = mx; __syncthreads();
  for (int st = 128; st > 0; st >>= 1) { if (tid < st) red[tid] = fmaxf(red[tid], red[tid + st]); __syncthreads(); }
  mx = red[0]; __syncthreads();
  float s = 0.f;
  for (int i = tid; i < n; i += 256) { float e = __expf(a[i] - mx); w[i] = e; s += e; }
  float t = blockReduceSum(s, red);
  for (int i = tid; i < n; i += 256) w[i] /= t;
}
__global__ __launch_bounds__(256) void sk_mean_vec(const float* a, float* out, int n) {
  __shared__ float red[256];
  float s = 0.f;
  for (int i = threadIdx.x; i < n; i += 256) s += a[i];
  float t = blockReduceSum(s, red);
  if (!threadIdx.x) out[0] = t / n;
}

// ------------------------------ WMMA GEMM ----------------------------------
// C[M,N](f32) = op(A)[M,K](bf16) * op(B)[K,N](bf16) (+bias)
// batching via element strides sA/sB/sC/sBias (gridDim.z = batch).
// B tile is staged in WMMA fragment order: Bf[tile][lane][16] so a lane's
// fragment is one contiguous 32B LDS read.
static constexpr int GBM = 128, GBN = 64, GBK = 32;

template <int TA, int TB>
__global__ __launch_bounds__(256)
void sk_gemm(const unsigned short* __restrict__ A, long long lda, long long sA,
             const unsigned short* __restrict__ B, long long ldb, long long sB,
             float* __restrict__ C, long long ldc, long long sC,
             const float* __restrict__ bias, long long sBias,
             int M, int N, int K) {
  __shared__ unsigned short As[GBM][GBK + 8];   // 80B rows (16B aligned)
  __shared__ unsigned short Bf[4][32][16];      // fragment-ordered B tile
  int bz = blockIdx.z;
  A += (long long)bz * sA; B += (long long)bz * sB; C += (long long)bz * sC;
  if (bias) bias += (long long)bz * sBias;
  int m0 = blockIdx.y * GBM, n0 = blockIdx.x * GBN;
  int tid = threadIdx.x, lane = tid & 31, wave = tid >> 5;
  int wm = (wave >> 1) * 32, wn = (wave & 1) * 32;
  const bool fullM = (m0 + GBM <= M);
  const bool fullN = (n0 + GBN <= N);
  v8f acc[2][2] = {};
  int ar = lane & 15, ko = (lane >> 4) << 3;
  for (int k0 = 0; k0 < K; k0 += GBK) {
    // ---- stage A (row-major [m][k] in LDS) ----
    if (fullM) {
      if (TA) {
        #pragma unroll
        for (int i = 0; i < 2; ++i) {
          int idx = tid + i * 256;                 // 512 chunks of 8
          int kk2 = idx >> 4, m8 = (idx & 15) * 8;
          V8U u; u.q = *(const uint4*)(A + (long long)(k0 + kk2) * lda + m0 + m8);
          #pragma unroll
          for (int t = 0; t < 8; ++t) As[m8 + t][kk2] = u.s[t];
        }
      } else {
        #pragma unroll
        for (int i = 0; i < 2; ++i) {
          int idx = tid + i * 256;
          int m = idx >> 2, c8 = (idx & 3) * 8;
          *(uint4*)&As[m][c8] = *(const uint4*)(A + (long long)(m0 + m) * lda + k0 + c8);
        }
      }
    } else {
      #pragma unroll
      for (int i = 0; i < (GBM * GBK) / 256; ++i) {
        int idx = tid + i * 256;
        int m = idx >> 5, kk2 = idx & 31;
        int gm = m0 + m, gk = k0 + kk2;
        unsigned short v = 0;
        if (gm < M) v = TA ? A[(long long)gk * lda + gm] : A[(long long)gm * lda + gk];
        As[m][kk2] = v;
      }
    }
    // ---- stage B (fragment order) ----
    if (fullN) {
      if (TB) {
        int nn = tid >> 2, k8 = (tid & 3) * 8;     // 64*4 = 256 chunks
        uint4 d = *(const uint4*)(B + (long long)(n0 + nn) * ldb + k0 + k8);
        *(uint4*)&Bf[nn >> 4][(nn & 15) + ((k8 >> 4) << 4)][k8 & 15] = d;
      } else {
        int kk2 = tid >> 3, n8 = (tid & 7) * 8;    // 32*8 = 256 chunks
        V8U u; u.q = *(const uint4*)(B + (long long)(k0 + kk2) * ldb + n0 + n8);
        int l2k = (kk2 >> 4) << 4, slot = kk2 & 15;
        #pragma unroll
        for (int t = 0; t < 8; ++t) {
          int nn = n8 + t;
          Bf[nn >> 4][(nn & 15) + l2k][slot] = u.s[t];
        }
      }
    } else {
      #pragma unroll
      for (int i = 0; i < (GBK * GBN) / 256; ++i) {
        int idx = tid + i * 256;
        int kk2 = idx >> 6, nn = idx & 63;
        int gn = n0 + nn, gk = k0 + kk2;
        unsigned short v = 0;
        if (gn < N) v = TB ? B[(long long)gn * ldb + gk] : B[(long long)gk * ldb + gn];
        Bf[nn >> 4][(nn & 15) + ((kk2 >> 4) << 4)][kk2 & 15] = v;
      }
    }
    __syncthreads();
    FragU fa[2], fb[2];
    #pragma unroll
    for (int i = 0; i < 2; ++i) {
      int row = wm + i * 16 + ar;
      fa[i].q[0] = *(const uint4*)&As[row][2 * ko];
      fa[i].q[1] = *(const uint4*)&As[row][2 * ko + 8];
    }
    #pragma unroll
    for (int j = 0; j < 2; ++j) {
      int jt = (wn >> 4) + j;
      fb[j].q[0] = *(const uint4*)&Bf[jt][lane][0];
      fb[j].q[1] = *(const uint4*)&Bf[jt][lane][8];
    }
    #pragma unroll
    for (int i = 0; i < 2; ++i)
      #pragma unroll
      for (int j = 0; j < 2; ++j)
        acc[i][j] = __builtin_amdgcn_wmma_f32_16x16x32_bf16(
            false, fa[i].v, false, fb[j].v, (short)0, acc[i][j], false, false);
    __syncthreads();
  }
  int rb = (lane >> 4) * 8, cc = lane & 15;
  #pragma unroll
  for (int i = 0; i < 2; ++i)
    #pragma unroll
    for (int j = 0; j < 2; ++j) {
      int col = n0 + wn + j * 16 + cc;
      if (col >= N) continue;
      float bv = bias ? bias[col] : 0.f;
      #pragma unroll
      for (int r = 0; r < 8; ++r) {
        int row = m0 + wm + i * 16 + rb + r;
        if (row < M) C[(long long)row * ldc + col] = acc[i][j][r] + bv;
      }
    }
}

// --------------------------- flash cross-attention -------------------------
// qh: (n, NH, HDM) bf16 (AI side), kh/vh: (n, NH, HDM) bf16 (user side)
// out ctx: (n, NH*HDM) f32. grid = (n/128, NH), block = 256 (8 waves x 16 rows)
__global__ __launch_bounds__(256)
void sk_flash(const unsigned short* __restrict__ Q, const unsigned short* __restrict__ Kd,
              const unsigned short* __restrict__ V, float* __restrict__ O,
              int n, float scale) {
  __shared__ unsigned short Kf[3][4][32][16];  // S B-fragments: [kchunk][ntile][lane][slot]
  __shared__ unsigned short Vf[2][6][32][16];  // PV B-fragments
  __shared__ unsigned short Ps[8][16][72];     // per-wave P tile, 144B rows
  int h = blockIdx.y;
  int qb = blockIdx.x * 128;
  int tid = threadIdx.x, lane = tid & 31, wave = tid >> 5;
  int ar = lane & 15, ko = (lane >> 4) << 3;
  int cc = lane & 15;
  // Q fragments straight from global (read once, aligned 16B chunks)
  FragU fq[3];
  {
    const unsigned short* qrow = Q + (long long)(qb + wave * 16 + ar) * DM + h * HDM;
    #pragma unroll
    for (int c = 0; c < 3; ++c) {
      fq[c].q[0] = *(const uint4*)(qrow + c * 32 + 2 * ko);
      fq[c].q[1] = *(const uint4*)(qrow + c * 32 + 2 * ko + 8);
    }
  }
  float mrow[8], lrow[8];
  #pragma unroll
  for (int r = 0; r < 8; ++r) { mrow[r] = -1e30f; lrow[r] = 0.f; }
  v8f o[6] = {};
  for (int kb = 0; kb < n; kb += 64) {
    // stage K tile in fragment order: element (key r, kdim c)
    #pragma unroll
    for (int i = 0; i < 3; ++i) {
      int idx = tid + i * 256;                  // 64 rows * 12 chunks
      int r = idx / 12, c8 = (idx % 12) * 8;
      uint4 d = *(const uint4*)(Kd + (long long)(kb + r) * DM + h * HDM + c8);
      *(uint4*)&Kf[c8 >> 5][r >> 4][(r & 15) + (((c8 >> 4) & 1) << 4)][c8 & 15] = d;
    }
    // stage V tile in fragment order: element (key r = k, col c = n)
    #pragma unroll
    for (int i = 0; i < 3; ++i) {
      int idx = tid + i * 256;
      int r = idx / 12, c8 = (idx % 12) * 8;
      V8U u; u.q = *(const uint4*)(V + (long long)(kb + r) * DM + h * HDM + c8);
      int c2 = r >> 5, l2k = ((r >> 4) & 1) << 4, slot = r & 15;
      #pragma unroll
      for (int t = 0; t < 8; ++t) {
        int nn = c8 + t;
        Vf[c2][nn >> 4][(nn & 15) + l2k][slot] = u.s[t];
      }
    }
    __syncthreads();
    // S = Q K^T (16 x 64 per wave)
    v8f s[4] = {};
    #pragma unroll
    for (int j = 0; j < 4; ++j)
      #pragma unroll
      for (int c = 0; c < 3; ++c) {
        FragU fk;
        fk.q[0] = *(const uint4*)&Kf[c][j][lane][0];
        fk.q[1] = *(const uint4*)&Kf[c][j][lane][8];
        s[j] = __builtin_amdgcn_wmma_f32_16x16x32_bf16(false, fq[c].v, false, fk.v, (short)0, s[j], false, false);
      }
    // online softmax per row (rows live in 16-lane halves)
    #pragma unroll
    for (int r = 0; r < 8; ++r) {
      float mx = -1e30f;
      #pragma unroll
      for (int j = 0; j < 4; ++j) { s[j][r] *= scale; mx = fmaxf(mx, s[j][r]); }
      #pragma unroll
      for (int off = 1; off < 16; off <<= 1) mx = fmaxf(mx, __shfl_xor(mx, off, 32));
      float mn = fmaxf(mrow[r], mx);
      float alpha = __expf(mrow[r] - mn);
      mrow[r] = mn;
      float rs = 0.f;
      #pragma unroll
      for (int j = 0; j < 4; ++j) { float p = __expf(s[j][r] - mn); s[j][r] = p; rs += p; }
      #pragma unroll
      for (int off = 1; off < 16; off <<= 1) rs += __shfl_xor(rs, off, 32);
      lrow[r] = lrow[r] * alpha + rs;
      #pragma unroll
      for (int t = 0; t < 6; ++t) o[t][r] *= alpha;
    }
    // stage P (per-wave region) and accumulate O += P*V
    int rb = (lane >> 4) * 8;
    #pragma unroll
    for (int j = 0; j < 4; ++j)
      #pragma unroll
      for (int r = 0; r < 8; ++r)
        Ps[wave][rb + r][j * 16 + cc] = f2bf(s[j][r]);
    #pragma unroll
    for (int c = 0; c < 2; ++c) {
      FragU fp;
      fp.q[0] = *(const uint4*)&Ps[wave][ar][c * 32 + 2 * ko];
      fp.q[1] = *(const uint4*)&Ps[wave][ar][c * 32 + 2 * ko + 8];
      #pragma unroll
      for (int t = 0; t < 6; ++t) {
        FragU fv;
        fv.q[0] = *(const uint4*)&Vf[c][t][lane][0];
        fv.q[1] = *(const uint4*)&Vf[c][t][lane][8];
        o[t] = __builtin_amdgcn_wmma_f32_16x16x32_bf16(false, fp.v, false, fv.v, (short)0, o[t], false, false);
      }
    }
    __syncthreads();
  }
  int rb = (lane >> 4) * 8;
  #pragma unroll
  for (int t = 0; t < 6; ++t) {
    int col = h * HDM + t * 16 + cc;
    #pragma unroll
    for (int r = 0; r < 8; ++r) {
      int row = qb + wave * 16 + rb + r;
      O[(long long)row * DM + col] = o[t][r] / lrow[r];
    }
  }
}

// ------------------------------ edge kernels -------------------------------
__global__ __launch_bounds__(256) void sk_edge_logits(const int* ei, int E, const float* q,
                                                      const float* kk, const float* p_rel,
                                                      float* logit, float* segmax) {
  int wid = blockIdx.x * 8 + (threadIdx.x >> 5);
  int lane = threadIdx.x & 31;
  if (wid >= E) return;
  int src = ei[wid], dst = ei[E + wid];
  const float scale = 0.1020620726159658f; // 1/sqrt(96)
  #pragma unroll
  for (int h = 0; h < NH; ++h) {
    float s = 0.f;
    for (int d = lane; d < HDM; d += 32)
      s += q[(long long)dst * DM + h * HDM + d] * kk[(long long)src * DM + h * HDM + d];
    for (int off = 16; off; off >>= 1) s += __shfl_xor(s, off, 32);
    s *= p_rel[h] * scale;
    if (!lane) { logit[(long long)wid * 4 + h] = s; atomicMaxF(&segmax[dst * 4 + h], s); }
  }
}
__global__ void sk_edge_expsum(const int* ei, int E, float* logit, const float* segmax, float* den) {
  long long i = (long long)blockIdx.x * 256 + threadIdx.x;
  if (i >= (long long)E * 4) return;
  int e = (int)(i >> 2), h = (int)(i & 3);
  int dst = ei[E + e];
  float w = __expf(logit[i] - segmax[dst * 4 + h]);
  logit[i] = w;
  atomicAdd(&den[dst * 4 + h], w);
}
__global__ __launch_bounds__(256) void sk_edge_scatter(const int* ei, int E, const float* logit,
                                                       const float* den, const float* vv, float* agg) {
  int wid = blockIdx.x * 8 + (threadIdx.x >> 5);
  int lane = threadIdx.x & 31;
  if (wid >= E) return;
  int src = ei[wid], dst = ei[E + wid];
  float wgt[4];
  #pragma unroll
  for (int h = 0; h < 4; ++h)
    wgt[h] = logit[(long long)wid * 4 + h] / (den[dst * 4 + h] + 1e-16f);
  #pragma unroll
  for (int j = 0; j < 12; ++j) {
    int idx = lane + 32 * j;
    int h = idx / HDM;
    atomicAdd(&agg[(long long)dst * DM + idx], wgt[h] * vv[(long long)src * DM + idx]);
  }
}
__global__ void sk_split(const float* kqv, unsigned short* kb, float* q, unsigned short* vb, int n) {
  long long i = (long long)blockIdx.x * 256 + threadIdx.x;
  if (i >= (long long)n * DM) return;
  long long r = i / DM; int c = (int)(i % DM);
  const float* row = kqv + r * D3;
  kb[i] = f2bf(row[c]);
  q[i]  = row[DM + c];
  vb[i] = f2bf(row[2 * DM + c]);
}

// ------------------------------ global block -------------------------------
__global__ __launch_bounds__(256) void sk_den(const unsigned short* qn, const float* kns, float* den, int n) {
  int wid = blockIdx.x * 8 + (threadIdx.x >> 5);
  int lane = threadIdx.x & 31;
  if (wid >= n * NH) return;
  int row = wid >> 2, h = wid & 3;
  float s = 0.f;
  for (int m = lane; m < DM; m += 32)
    s += bf2f(qn[(long long)row * DHG + h * DM + m]) * kns[h * DM + m];
  for (int off = 16; off; off >>= 1) s += __shfl_xor(s, off, 32);
  if (!lane) den[row * 4 + h] = s + (float)n;
}
__global__ void sk_gcomb(const float* num, const float* vvf, const float* den4,
                         const float* x0, float* out, int n) {
  long long i = (long long)blockIdx.x * 256 + threadIdx.x;
  if (i >= (long long)n * DM) return;
  long long r = i / DM; int c = (int)(i % DM);
  float acc = 0.f;
  #pragma unroll
  for (int h = 0; h < NH; ++h)
    acc += (num[r * DHG + h * DM + c] + (float)n * vvf[r * DHG + h * DM + c]) / den4[r * 4 + h];
  out[i] = 0.5f * (acc * 0.25f) + 0.5f * x0[i];
}

// ------------------------------ misc tail ----------------------------------
__global__ void sk_concat_bf(const float* a, const float* b, unsigned short* dst, int rows) {
  long long i = (long long)blockIdx.x * 256 + threadIdx.x;
  int cols = 2 * DM;
  if (i >= (long long)rows * cols) return;
  long long r = i / cols; int c = (int)(i % cols);
  float v = (c < DM) ? a[r * DM + c] : b[r * DM + (c - DM)];
  dst[i] = f2bf(v);
}
__global__ __launch_bounds__(256) void sk_rowdot_sig(const float* X, const float* w, const float* bp,
                                                     float* out, int rows, int K) {
  int wid = blockIdx.x * 8 + (threadIdx.x >> 5);
  int lane = threadIdx.x & 31;
  if (wid >= rows) return;
  float s = 0.f;
  for (int k = lane; k < K; k += 32) s += X[(long long)wid * K + k] * w[k];
  for (int off = 16; off; off >>= 1) s += __shfl_xor(s, off, 32);
  if (!lane) out[wid] = sigf(s + bp[0]);
}
__global__ __launch_bounds__(256) void sk_gemv(const float* x, const float* W, const float* b,
                                               float* out, int K, int Nout, int act) {
  __shared__ float red[256];
  int j = blockIdx.x;
  if (j >= Nout) return;
  float s = 0.f;
  for (int k = threadIdx.x; k < K; k += 256) s += x[k] * W[(long long)k * Nout + j];
  float t = blockReduceSum(s, red);
  if (!threadIdx.x) {
    float v = t + b[j];
    if (act == 1) v = fmaxf(v, 0.f);
    out[j] = v;
  }
}
__global__ void sk_build_comb(const float* upool, const float* apool, const float* aalign,
                              const float* meanp, float* comb) {
  int i = blockIdx.x * 256 + threadIdx.x;
  if (i >= 4 * DM) return;
  float v;
  if (i < DM) v = upool[i];
  else if (i < 2 * DM) v = apool[i - DM];
  else if (i < 3 * DM) v = aalign[i - 2 * DM];
  else v = *meanp;
  comb[i] = v;
}
__global__ void sk_pack(float* out, const float* lgs, const float* align_, const float* turn_, int n) {
  int i = blockIdx.x * 256 + threadIdx.x;
  if (i == 0) { out[0] = lgs[0]; out[1] = sigf(lgs[0]); }
  if (i < n) { out[2 + i] = align_[i]; out[2 + n + i] = turn_[i]; }
}

// ---------------------------------------------------------------------------
extern "C" void kernel_launch(void* const* d_in, const int* in_sizes, int n_in,
                              void* d_out, int out_size, void* d_ws, size_t ws_size,
                              hipStream_t stream) {
  (void)n_in; (void)out_size; (void)ws_size;
  const int N = in_sizes[0] / DM;    // 4096
  const int E = in_sizes[2] / 2;     // 65536
  auto F = [&](int i) { return (const float*)d_in[i]; };
  const float* user_x = F(0);
  const float* ai_x   = F(1);
  const int* edg[4] = { (const int*)d_in[2], (const int*)d_in[3],
                        (const int*)d_in[4], (const int*)d_in[5] };
  const int et_src[4] = { 0, 1, 1, 0 };   // ua, au, aa, uu (0=user,1=ai)
  const int et_dst[4] = { 1, 0, 1, 0 };
  // params flattened in setup_inputs() insertion order:
  // hgt0 (26 leaves), hgt1 (26), global (15), inter (16), head (14)
  const int PB = 6;
  auto hp = [&](int l, int o) { return F(PB + l * 26 + o); };
  const int GB = PB + 52;   // global base
  const int IB = GB + 15;   // inter base
  const int HB = IB + 16;   // head base

  size_t off = 0;
  auto alc = [&](size_t bytes) -> char* {
    char* p = (char*)d_ws + off;
    off = (off + bytes + 255) & ~(size_t)255;
    return p;
  };
  auto AFp = [&](size_t n) { return (float*)alc(n * sizeof(float)); };
  auto AUp = [&](size_t n) { return (unsigned short*)alc(n * sizeof(unsigned short)); };
  auto EWG = [&](long long n) { return dim3((unsigned)((n + 255) / 256)); };

  // ---- persistent buffers ----
  float* xu   = AFp((size_t)N * DM);
  float* xa   = AFp((size_t)N * DM);
  float* ai_g = AFp((size_t)N * DM);
  float* ctx  = AFp((size_t)N * DM);
  float* alig = AFp(N);
  float* turn = AFp(N);
  size_t scratch0 = off;

  auto tobf = [&](const float* s, unsigned short* d, long long n) {
    sk_f2bf<<<EWG(n), 256, 0, stream>>>(s, d, n);
  };
  auto fillf = [&](float* p, float v, long long n) {
    sk_fill<<<EWG(n), 256, 0, stream>>>(p, v, n);
  };
  auto gemm = [&](const unsigned short* A, long long lda, long long sA,
                  const unsigned short* B, long long ldb, long long sB,
                  float* C, long long ldc, long long sC,
                  const float* bias, long long sBias,
                  int M, int Nn, int K, int tA, int tB, int batch) {
    dim3 g((Nn + GBN - 1) / GBN, (M + GBM - 1) / GBM, batch);
    if (tA == 0 && tB == 0)
      sk_gemm<0, 0><<<g, 256, 0, stream>>>(A, lda, sA, B, ldb, sB, C, ldc, sC, bias, sBias, M, Nn, K);
    else if (tA == 1)
      sk_gemm<1, 0><<<g, 256, 0, stream>>>(A, lda, sA, B, ldb, sB, C, ldc, sC, bias, sBias, M, Nn, K);
    else
      sk_gemm<0, 1><<<g, 256, 0, stream>>>(A, lda, sA, B, ldb, sB, C, ldc, sC, bias, sBias, M, Nn, K);
  };

  sk_copyf<<<EWG((long long)N * DM), 256, 0, stream>>>(user_x, xu, (long long)N * DM);
  sk_copyf<<<EWG((long long)N * DM), 256, 0, stream>>>(ai_x, xa, (long long)N * DM);

  // ========================== HGT layers =================================
  off = scratch0;
  unsigned short* xbf[2]  = { AUp((size_t)N * DM), AUp((size_t)N * DM) };
  float* kqvb             = AFp((size_t)N * D3);
  unsigned short* kbf[2]  = { AUp((size_t)N * DM), AUp((size_t)N * DM) };
  float* qfl[2]           = { AFp((size_t)N * DM), AFp((size_t)N * DM) };
  unsigned short* vbf[2]  = { AUp((size_t)N * DM), AUp((size_t)N * DM) };
  float* kkb              = AFp((size_t)N * DM);
  float* vvb[4]           = { AFp((size_t)N * DM), AFp((size_t)N * DM),
                              AFp((size_t)N * DM), AFp((size_t)N * DM) };
  float* logit[2]         = { AFp((size_t)2 * E * 4), AFp((size_t)2 * E * 4) };
  float* smax[2]          = { AFp((size_t)N * 4), AFp((size_t)N * 4) };
  float* sden[2]          = { AFp((size_t)N * 4), AFp((size_t)N * 4) };
  float* agg[2]           = { AFp((size_t)N * DM), AFp((size_t)N * DM) };
  unsigned short* aggbf   = AUp((size_t)N * DM);
  float* obuf             = AFp((size_t)N * DM);
  float* tmpf             = AFp((size_t)N * DM);
  unsigned short* wbig    = AUp((size_t)DM * D3);
  unsigned short* wrel    = AUp((size_t)NH * HDM * HDM);
  float* X[2] = { xu, xa };

  for (int l = 0; l < 2; ++l) {
    tobf(xu, xbf[0], (long long)N * DM);
    tobf(xa, xbf[1], (long long)N * DM);
    for (int t = 0; t < 2; ++t) {
      tobf(hp(l, t * 7 + 0), wbig, (long long)DM * D3);
      gemm(xbf[t], DM, 0, wbig, D3, 0, kqvb, D3, 0, hp(l, t * 7 + 1), 0, N, D3, DM, 0, 0, 1);
      sk_split<<<EWG((long long)N * DM), 256, 0, stream>>>(kqvb, kbf[t], qfl[t], vbf[t], N);
    }
    for (int t = 0; t < 2; ++t) {
      fillf(smax[t], -1e30f, N * 4);
      fillf(sden[t], 0.f, N * 4);
      fillf(agg[t], 0.f, (long long)N * DM);
    }
    float* lgp[4]; int slot[2] = { 0, 0 };
    for (int e = 0; e < 4; ++e) {
      int s = et_src[e], d = et_dst[e];
      tobf(hp(l, 14 + e * 3), wrel, (long long)NH * HDM * HDM);
      gemm(kbf[s], DM, HDM, wrel, HDM, HDM * HDM, kkb, DM, HDM, nullptr, 0, N, HDM, HDM, 0, 0, NH);
      lgp[e] = logit[d] + (size_t)slot[d] * E * 4; slot[d]++;
      sk_edge_logits<<<dim3((E + 7) / 8), 256, 0, stream>>>(edg[e], E, qfl[d], kkb,
                                                            hp(l, 16 + e * 3), lgp[e], smax[d]);
      tobf(hp(l, 15 + e * 3), wrel, (long long)NH * HDM * HDM);
      gemm(vbf[s], DM, HDM, wrel, HDM, HDM * HDM, vvb[e], DM, HDM, nullptr, 0, N, HDM, HDM, 0, 0, NH);
    }
    for (int e = 0; e < 4; ++e)
      sk_edge_expsum<<<EWG((long long)E * 4), 256, 0, stream>>>(edg[e], E, lgp[e],
                                                                smax[et_dst[e]], sden[et_dst[e]]);
    for (int e = 0; e < 4; ++e)
      sk_edge_scatter<<<dim3((E + 7) / 8), 256, 0, stream>>>(edg[e], E, lgp[e],
                                                             sden[et_dst[e]], vvb[e], agg[et_dst[e]]);
    for (int t = 0; t < 2; ++t) {
      sk_gelu2bf<<<EWG((long long)N * DM), 256, 0, stream>>>(agg[t], aggbf, (long long)N * DM);
      tobf(hp(l, t * 7 + 2), wbig, (long long)DM * DM);
      gemm(aggbf, DM, 0, wbig, DM, 0, obuf, DM, 0, hp(l, t * 7 + 3), 0, N, DM, DM, 0, 0, 1);
      sk_skip<<<EWG((long long)N * DM), 256, 0, stream>>>(X[t], obuf, hp(l, t * 7 + 4), (long long)N * DM);
      sk_gelu<<<EWG((long long)N * DM), 256, 0, stream>>>(X[t], tmpf, (long long)N * DM);
      sk_layernorm<<<N, 256, 0, stream>>>(tmpf, hp(l, t * 7 + 5), hp(l, t * 7 + 6), X[t]);
    }
  }

  // ========================== global block ===============================
  off = scratch0;
  unsigned short* gxbf  = AUp((size_t)N * DM);
  float* fco            = AFp((size_t)N * DM);
  float* x0             = AFp((size_t)N * DM);
  unsigned short* x0bf  = AUp((size_t)N * DM);
  float* gq             = AFp((size_t)N * DHG);
  float* gk             = AFp((size_t)N * DHG);
  float* gv             = AFp((size_t)N * DHG);
  unsigned short* qnbf  = AUp((size_t)N * DHG);
  unsigned short* knbf  = AUp((size_t)N * DHG);
  unsigned short* gvbf  = AUp((size_t)N * DHG);
  float* kns            = AFp(DHG);
  float* kvs            = AFp((size_t)NH * DM * DM);
  unsigned short* kvsbf = AUp((size_t)NH * DM * DM);
  float* numb           = AFp((size_t)N * DHG);
  float* den4           = AFp((size_t)N * NH);
  float* xmid           = AFp((size_t)N * DM);
  float* x1             = AFp((size_t)N * DM);
  float* ssq            = AFp(8);
  unsigned short* wg    = AUp((size_t)DM * DHG);

  tobf(xa, gxbf, (long long)N * DM);
  tobf(F(GB + 0), wg, (long long)DM * DM);
  gemm(gxbf, DM, 0, wg, DM, 0, fco, DM, 0, F(GB + 1), 0, N, DM, DM, 0, 0, 1);
  sk_layernorm<<<N, 256, 0, stream>>>(fco, F(GB + 2), F(GB + 3), x0);
  sk_relu<<<EWG((long long)N * DM), 256, 0, stream>>>(x0, x0, (long long)N * DM);
  tobf(x0, x0bf, (long long)N * DM);
  tobf(F(GB + 4), wg, (long long)DM * DHG);
  gemm(x0bf, DM, 0, wg, DHG, 0, gq, DHG, 0, F(GB + 5), 0, N, DHG, DM, 0, 0, 1);
  tobf(F(GB + 6), wg, (long long)DM * DHG);
  gemm(x0bf, DM, 0, wg, DHG, 0, gk, DHG, 0, F(GB + 7), 0, N, DHG, DM, 0, 0, 1);
  tobf(F(GB + 8), wg, (long long)DM * DHG);
  gemm(x0bf, DM, 0, wg, DHG, 0, gv, DHG, 0, F(GB + 9), 0, N, DHG, DM, 0, 0, 1);
  fillf(ssq, 0.f, 2);
  sk_sumsq<<<256, 256, 0, stream>>>(gq, ssq + 0, (long long)N * DHG);
  sk_sumsq<<<256, 256, 0, stream>>>(gk, ssq + 1, (long long)N * DHG);
  sk_scalebf<<<EWG((long long)N * DHG), 256, 0, stream>>>(gq, ssq + 0, qnbf, (long long)N * DHG);
  sk_scalebf<<<EWG((long long)N * DHG), 256, 0, stream>>>(gk, ssq + 1, knbf, (long long)N * DHG);
  tobf(gv, gvbf, (long long)N * DHG);
  sk_colsum_bf<<<DHG, 256, 0, stream>>>(knbf, kns, N, DHG);
  // kvs[h] = kn[:,h,:]^T @ vv[:,h,:]  (transA, K = N)
  gemm(knbf, DHG, DM, gvbf, DHG, DM, kvs, DM, (long long)DM * DM, nullptr, 0, DM, DM, N, 1, 0, NH);
  tobf(kvs, kvsbf, (long long)NH * DM * DM);
  gemm(qnbf, DHG, DM, kvsbf, DM, (long long)DM * DM, numb, DHG, DM, nullptr, 0, N, DM, DM, 0, 0, NH);
  sk_den<<<dim3((N * NH + 7) / 8), 256, 0, stream>>>(qnbf, kns, den4, N);
  sk_gcomb<<<EWG((long long)N * DM), 256, 0, stream>>>(numb, gv, den4, x0, xmid, N);
  sk_layernorm<<<N, 256, 0, stream>>>(xmid, F(GB + 10), F(GB + 11), x1);
  sk_mix<<<EWG((long long)N * DM), 256, 0, stream>>>(xa, x1, F(GB + 14), xmid, (long long)N * DM);
  sk_layernorm<<<N, 256, 0, stream>>>(xmid, F(GB + 12), F(GB + 13), ai_g);

  // ========================== inter block ================================
  off = scratch0;
  unsigned short* ibf   = AUp((size_t)N * DM);
  float* t0             = AFp((size_t)N * DM);
  float* upb_           = AFp((size_t)N * DM);
  float* apb_           = AFp((size_t)N * DM);
  unsigned short* upbf  = AUp((size_t)N * DM);
  unsigned short* apbf  = AUp((size_t)N * DM);
  float* qhf            = AFp((size_t)N * DM);
  float* khf            = AFp((size_t)N * DM);
  float* vhf            = AFp((size_t)N * DM);
  unsigned short* qhb   = AUp((size_t)N * DM);
  unsigned short* khb   = AUp((size_t)N * DM);
  unsigned short* vhb   = AUp((size_t)N * DM);
  float* ctxa           = AFp((size_t)N * DM);
  unsigned short* ctxab = AUp((size_t)N * DM);
  unsigned short* combbf = AUp((size_t)N * 2 * DM);
  float* al1o           = AFp((size_t)N * DM);
  unsigned short* wib   = AUp((size_t)3 * DM * DM);

  tobf(xu, ibf, (long long)N * DM);
  tobf(F(IB + 0), wib, (long long)DM * DM);
  gemm(ibf, DM, 0, wib, DM, 0, t0, DM, 0, F(IB + 1), 0, N, DM, DM, 0, 0, 1);
  sk_layernorm<<<N, 256, 0, stream>>>(t0, F(IB + 2), F(IB + 3), upb_);
  sk_gelu<<<EWG((long long)N * DM), 256, 0, stream>>>(upb_, upb_, (long long)N * DM);
  tobf(ai_g, ibf, (long long)N * DM);
  tobf(F(IB + 4), wib, (long long)DM * DM);
  gemm(ibf, DM, 0, wib, DM, 0, t0, DM, 0, F(IB + 5), 0, N, DM, DM, 0, 0, 1);
  sk_layernorm<<<N, 256, 0, stream>>>(t0, F(IB + 6), F(IB + 7), apb_);
  sk_gelu<<<EWG((long long)N * DM), 256, 0, stream>>>(apb_, apb_, (long long)N * DM);
  tobf(upb_, upbf, (long long)N * DM);
  tobf(apb_, apbf, (long long)N * DM);
  tobf(F(IB + 8), wib, (long long)3 * DM * DM);   // in_w (1152,384)
  const float* in_b = F(IB + 9);
  gemm(apbf, DM, 0, wib, DM, 0, qhf, DM, 0, in_b, 0, N, DM, DM, 0, 1, 1);                 // ap @ wq.T
  gemm(upbf, DM, 0, wib + DM * DM, DM, 0, khf, DM, 0, in_b + DM, 0, N, DM, DM, 0, 1, 1);  // up @ wk.T
  gemm(upbf, DM, 0, wib + 2 * DM * DM, DM, 0, vhf, DM, 0, in_b + 2 * DM, 0, N, DM, DM, 0, 1, 1);
  tobf(qhf, qhb, (long long)N * DM);
  tobf(khf, khb, (long long)N * DM);
  tobf(vhf, vhb, (long long)N * DM);
  sk_flash<<<dim3(N / 128, NH), 256, 0, stream>>>(qhb, khb, vhb, ctxa, N, 0.1020620726159658f);
  tobf(ctxa, ctxab, (long long)N * DM);
  tobf(F(IB + 10), wib, (long long)DM * DM);
  gemm(ctxab, DM, 0, wib, DM, 0, ctx, DM, 0, F(IB + 11), 0, N, DM, DM, 0, 1, 1);          // ctx @ o_w.T
  sk_concat_bf<<<EWG((long long)N * 2 * DM), 256, 0, stream>>>(apb_, ctx, combbf, N);
  tobf(F(IB + 12), wib, (long long)2 * DM * DM);
  gemm(combbf, 2 * DM, 0, wib, DM, 0, al1o, DM, 0, F(IB + 13), 0, N, DM, 2 * DM, 0, 0, 1);
  sk_relu<<<EWG((long long)N * DM), 256, 0, stream>>>(al1o, al1o, (long long)N * DM);
  sk_rowdot_sig<<<dim3((N + 7) / 8), 256, 0, stream>>>(al1o, F(IB + 14), F(IB + 15), alig, N, DM);

  // ============================== head ===================================
  off = scratch0;
  float* aw    = AFp(N);
  float* tAv   = AFp(DM);
  float* tBv   = AFp(DM);
  float* upool = AFp(DM);
  float* apool = AFp(DM);
  float* aalign = AFp(DM);
  float* meanv = AFp(4);
  float* comb  = AFp(4 * DM);
  float* h0    = AFp(DM);
  float* h1    = AFp(DM);
  float* h2    = AFp(DM / 2);
  float* lgs   = AFp(4);

  sk_colmean<<<DM, 256, 0, stream>>>(xu, tAv, N, DM);
  sk_layernorm<<<1, 256, 0, stream>>>(tAv, F(HB + 0), F(HB + 1), upool);
  sk_colmean<<<DM, 256, 0, stream>>>(ai_g, tBv, N, DM);
  sk_layernorm<<<1, 256, 0, stream>>>(tBv, F(HB + 2), F(HB + 3), apool);
  sk_softmax_vec<<<1, 256, 0, stream>>>(alig, aw, N);
  sk_colwsum<<<DM, 256, 0, stream>>>(ctx, aw, aalign, N, DM);
  sk_mean_vec<<<1, 256, 0, stream>>>(alig, meanv, N);
  sk_build_comb<<<6, 256, 0, stream>>>(upool, apool, aalign, meanv, comb);
  sk_gemv<<<DM, 256, 0, stream>>>(comb, F(HB + 4), F(HB + 5), h0, 4 * DM, DM, 0);
  sk_layernorm<<<1, 256, 0, stream>>>(h0, F(HB + 6), F(HB + 7), h1);
  sk_relu<<<EWG(DM), 256, 0, stream>>>(h1, h1, DM);
  sk_gemv<<<DM / 2, 256, 0, stream>>>(h1, F(HB + 8), F(HB + 9), h2, DM, DM / 2, 1);
  sk_gemv<<<1, 256, 0, stream>>>(h2, F(HB + 10), F(HB + 11), lgs, DM / 2, 1, 0);
  sk_rowdot_sig<<<dim3((N + 7) / 8), 256, 0, stream>>>(ctx, F(HB + 12), F(HB + 13), turn, N, DM);
  sk_pack<<<EWG(N), 256, 0, stream>>>((float*)d_out, lgs, alig, turn, N);
}
